// ViTFVLM_32203664786005
// MI455X (gfx1250) — matrix operand order repaired and checked by reading, not compiled, not run on hardware
//
#include <hip/hip_runtime.h>

// ---------------------------------------------------------------------------
// ViT-Base forward for MI455X (gfx1250, wave32, WMMA).
// - All matmuls (patch embed, QKV, attention QK^T/PV, out-proj, MLP) via
//   v_wmma_f32_16x16x32_bf16 with f32 accumulation.
// - All WMMA operands live in memory as bf16; weights converted+transposed to
//   [N][K] once per call so every tile is a row-contiguous async B128 copy.
// - Tile staging uses GLOBAL_LOAD_ASYNC_TO_LDS_B128 (+ s_wait_asynccnt) when
//   the toolchain exposes it; otherwise a plain vector copy.
// ---------------------------------------------------------------------------

#define NT      1232          // tokens per batch (7*16*11)
#define BATCH   2
#define MROWS   (BATCH*NT)    // 2464
#define EDIM    768
#define HEADS   12
#define DHEAD   64
#define QKVD    2304
#define MLPD    3072
#define KPATCH  4096          // 16*16*16
#define NLAYER  12
#define EPSLN   1e-5f
#define ATT_SCALE 0.125f      // 64^-0.5

typedef unsigned short bf16_t;
typedef __attribute__((ext_vector_type(16))) __bf16 v16bf;
typedef __attribute__((ext_vector_type(8)))  float  v8f;

union FragAB { v16bf v; unsigned u[8]; };

#if defined(__AMDGCN__)
#  if __has_builtin(__builtin_amdgcn_global_load_async_to_lds_b128)
#    define ASYNC_CP 1
#  endif
#endif

__device__ inline v8f vzero() {
  v8f z;
#pragma unroll
  for (int i = 0; i < 8; i++) z[i] = 0.f;
  return z;
}

// fp32 -> bf16 round-to-nearest-even
__device__ inline bf16_t f2bf(float f) {
  unsigned u = __builtin_bit_cast(unsigned, f);
  unsigned r = u + 0x7FFFu + ((u >> 16) & 1u);
  return (bf16_t)(r >> 16);
}

__device__ inline v8f wmma_bf16(const FragAB& a, const FragAB& b, v8f c) {
  return __builtin_amdgcn_wmma_f32_16x16x32_bf16(
      false, a.v, false, b.v, (short)0, c, false, false);
}

// 16-byte global -> LDS copy (async on gfx1250 when available)
#if defined(ASYNC_CP)
typedef int avec4 __attribute__((vector_size(16)));
typedef __attribute__((address_space(1))) avec4* g_v4i_ptr;
typedef __attribute__((address_space(3))) avec4* l_v4i_ptr;
#endif

__device__ inline void cp16(const bf16_t* __restrict__ g, bf16_t* __restrict__ l) {
#if defined(ASYNC_CP)
  void* gv = (void*)g;
  void* lv = (void*)l;
  __builtin_amdgcn_global_load_async_to_lds_b128((g_v4i_ptr)gv, (l_v4i_ptr)lv,
                                                 0, 0);
#else
  *(uint4*)l = *(const uint4*)g;
#endif
}

__device__ inline void cp_wait() {
#if defined(ASYNC_CP)
#  if __has_builtin(__builtin_amdgcn_s_wait_asynccnt)
  __builtin_amdgcn_s_wait_asynccnt(0);
#  else
  asm volatile("s_wait_asynccnt 0" ::: "memory");
#  endif
#endif
}

__device__ inline void storeOut(float* p, float v) { *p = v; }
__device__ inline void storeOut(bf16_t* p, float v) { *p = f2bf(v); }

// ---------------------------------------------------------------------------
// GEMM: C[M,N] = A[M,K](bf16) @ Bt[N,K](bf16, pre-transposed) + bias (+epi)
// EPI: 0 = bias, 1 = bias + exact GELU, 2 = bias + residual add (fp32 resid)
// 128x128x32 tile, 256 threads = 8 waves (2x4), wave tile 64x32.
// ---------------------------------------------------------------------------
template <int EPI, typename OutT>
__global__ __launch_bounds__(256) void gemm_k(
    const bf16_t* __restrict__ A, const bf16_t* __restrict__ Bt,
    const float* __restrict__ bias, const float* __restrict__ resid,
    OutT* __restrict__ C, int M, int N, int K) {
  __shared__ __align__(16) bf16_t As[128 * 40];  // [m][k] pad->40 (80B rows)
  __shared__ __align__(16) bf16_t Bs[128 * 40];  // [n][k]

  const int t = threadIdx.x;
  const int wave = t >> 5, lane = t & 31;
  const int wr = wave >> 2, wc = wave & 3;
  const int ln = lane & 15, hf = lane >> 4;
  const int m0 = blockIdx.y * 128, n0 = blockIdx.x * 128;

  v8f acc[4][2];
#pragma unroll
  for (int r = 0; r < 4; r++)
#pragma unroll
    for (int c = 0; c < 2; c++) acc[r][c] = vzero();

  for (int k0 = 0; k0 < K; k0 += 32) {
    __syncthreads();
    // A tile: 128 rows x 64B (clamp edge rows; stores are guarded later)
#pragma unroll
    for (int i = 0; i < 2; i++) {
      int idx = i * 256 + t;      // 0..511
      int m = idx >> 2, c4 = idx & 3;
      int gm = m0 + m; if (gm >= M) gm = M - 1;
      cp16(A + (size_t)gm * K + k0 + c4 * 8, &As[m * 40 + c4 * 8]);
    }
    // B tile: 128 rows x 64B from [N][K]
#pragma unroll
    for (int i = 0; i < 2; i++) {
      int idx = i * 256 + t;
      int n = idx >> 2, c4 = idx & 3;
      cp16(Bt + (size_t)(n0 + n) * K + k0 + c4 * 8, &Bs[n * 40 + c4 * 8]);
    }
    cp_wait();
    __syncthreads();

    FragAB a[4], b[2];
#pragma unroll
    for (int r = 0; r < 4; r++) {
      int row = wr * 64 + r * 16 + ln;
#pragma unroll
      for (int v = 0; v < 8; v++) {
        int kb = (v & 3) * 2 + hf * 8 + (v >> 2) * 16;
        a[r].u[v] = *(const unsigned*)&As[row * 40 + kb];
      }
    }
#pragma unroll
    for (int c = 0; c < 2; c++) {
      int n = wc * 32 + c * 16 + ln;
#pragma unroll
      for (int v = 0; v < 8; v++) {
        int kb = (v & 3) * 2 + hf * 8 + (v >> 2) * 16;
        b[c].u[v] = *(const unsigned*)&Bs[n * 40 + kb];
      }
    }
#pragma unroll
    for (int r = 0; r < 4; r++)
#pragma unroll
      for (int c = 0; c < 2; c++) acc[r][c] = wmma_bf16(a[r], b[c], acc[r][c]);
  }

#pragma unroll
  for (int r = 0; r < 4; r++)
#pragma unroll
    for (int c = 0; c < 2; c++)
#pragma unroll
      for (int e = 0; e < 8; e++) {
        int gm = m0 + wr * 64 + r * 16 + e + 8 * hf;
        int gn = n0 + wc * 32 + c * 16 + ln;
        if (gm < M) {
          float v = acc[r][c][e] + bias[gn];
          if (EPI == 1) v = 0.5f * v * (1.0f + erff(v * 0.70710678f));
          if (EPI == 2) v += resid[(size_t)gm * N + gn];
          storeOut(&C[(size_t)gm * N + gn], v);
        }
      }
}

// ---------------------------------------------------------------------------
// Fused flash attention per (batch, head). 128 threads = 4 waves; wave owns
// 16 query rows; 64-key tiles; online softmax; QK^T and PV via WMMA.
// qkv: bf16 [B, NT, 3, H, DH]. Output: bf16 [B*NT, 768].
// ---------------------------------------------------------------------------
__global__ __launch_bounds__(128) void attn_k(const bf16_t* __restrict__ qkvb,
                                              bf16_t* __restrict__ outp) {
  constexpr int ST = 72;  // padded row stride (144B, 16B aligned)
  __shared__ __align__(16) bf16_t Qs[64 * ST];  // [q][dh]
  __shared__ __align__(16) bf16_t Ks[64 * ST];  // [key][dh]
  __shared__ __align__(16) bf16_t Vs[64 * ST];  // [dh][key] (transposed)
  __shared__ __align__(16) bf16_t Ps[64 * ST];  // [q][key], per-wave rows

  const int t = threadIdx.x;
  const int wave = t >> 5, lane = t & 31;
  const int ln = lane & 15, hf = lane >> 4;
  const int bh = blockIdx.y;
  const int b = bh / HEADS, h = bh % HEADS;
  const int q0 = blockIdx.x * 64;

  // stage Q once: 64 rows x 128B, async B128 copies
#pragma unroll
  for (int i = 0; i < 4; i++) {
    int idx = i * 128 + t;      // 0..511
    int m = idx >> 3, c8 = idx & 7;
    int row = q0 + m; if (row >= NT) row = NT - 1;
    cp16(qkvb + ((size_t)(b * NT + row) * 3 + 0) * 768 + h * DHEAD + c8 * 8,
         &Qs[m * ST + c8 * 8]);
  }

  float m_run[8], l_run[8];
#pragma unroll
  for (int e = 0; e < 8; e++) { m_run[e] = -1e30f; l_run[e] = 0.f; }
  v8f o[4];
#pragma unroll
  for (int c = 0; c < 4; c++) o[c] = vzero();

  const int nkt = (NT + 63) / 64;
  for (int kt = 0; kt < nkt; kt++) {
    __syncthreads();
    // K tile async; rows past NT are clamped (masked in softmax)
#pragma unroll
    for (int i = 0; i < 4; i++) {
      int idx = i * 128 + t;
      int key = idx >> 3, c8 = idx & 7;
      int row = kt * 64 + key; if (row >= NT) row = NT - 1;
      cp16(qkvb + ((size_t)(b * NT + row) * 3 + 1) * 768 + h * DHEAD + c8 * 8,
           &Ks[key * ST + c8 * 8]);
    }
    // V tile, transposed manually (zero for masked keys; P is 0 there anyway)
#pragma unroll
    for (int i = 0; i < 32; i++) {
      int idx = t * 32 + i;
      int key = idx >> 6, d = idx & 63;
      int row = kt * 64 + key;
      Vs[d * ST + key] =
          (row < NT)
              ? qkvb[((size_t)(b * NT + row) * 3 + 2) * 768 + h * DHEAD + d]
              : (bf16_t)0;
    }
    cp_wait();
    __syncthreads();

    // S = Q K^T : load all fragments first, then WMMA burst
    v8f s[4];
#pragma unroll
    for (int c = 0; c < 4; c++) s[c] = vzero();
    {
      FragAB aq[2], bk[2][4];
      int rowA = wave * 16 + ln;
#pragma unroll
      for (int kh = 0; kh < 2; kh++) {
        int kk = kh * 32;
#pragma unroll
        for (int v = 0; v < 8; v++) {
          int kb = (v & 3) * 2 + hf * 8 + (v >> 2) * 16 + kk;
          aq[kh].u[v] = *(const unsigned*)&Qs[rowA * ST + kb];
        }
#pragma unroll
        for (int c = 0; c < 4; c++) {
          int nk = c * 16 + ln;
#pragma unroll
          for (int v = 0; v < 8; v++) {
            int kb = (v & 3) * 2 + hf * 8 + (v >> 2) * 16 + kk;
            bk[kh][c].u[v] = *(const unsigned*)&Ks[nk * ST + kb];
          }
        }
      }
#pragma unroll
      for (int kh = 0; kh < 2; kh++)
#pragma unroll
        for (int c = 0; c < 4; c++) s[c] = wmma_bf16(aq[kh], bk[kh][c], s[c]);
    }

    // online softmax (per-lane stats cover 8 rows of this half)
#pragma unroll
    for (int e = 0; e < 8; e++) {
      float mx = -1e30f;
#pragma unroll
      for (int c = 0; c < 4; c++) {
        float sv = s[c][e] * ATT_SCALE;
        int kg = kt * 64 + c * 16 + ln;
        if (kg >= NT) sv = -1e30f;
        s[c][e] = sv;
        mx = fmaxf(mx, sv);
      }
#pragma unroll
      for (int off = 1; off < 16; off <<= 1)
        mx = fmaxf(mx, __shfl_xor(mx, off, 32));
      float newm = fmaxf(m_run[e], mx);
      float sc = __expf(m_run[e] - newm);
      m_run[e] = newm;
      float rs = 0.f;
#pragma unroll
      for (int c = 0; c < 4; c++) {
        float p = __expf(s[c][e] - newm);
        s[c][e] = p;
        rs += p;
      }
#pragma unroll
      for (int off = 1; off < 16; off <<= 1) rs += __shfl_xor(rs, off, 32);
      l_run[e] = l_run[e] * sc + rs;
#pragma unroll
      for (int c = 0; c < 4; c++) o[c][e] *= sc;
    }

    // P -> LDS (wave-private rows), re-fragment as A
#pragma unroll
    for (int c = 0; c < 4; c++)
#pragma unroll
      for (int e = 0; e < 8; e++) {
        int m = e + 8 * hf;
        Ps[(wave * 16 + m) * ST + c * 16 + ln] = f2bf(s[c][e]);
      }

    // O += P @ V, all fragments hoisted
    {
      FragAB ap[2], bv[2][4];
      int rowA = wave * 16 + ln;
#pragma unroll
      for (int kh = 0; kh < 2; kh++) {
        int kk = kh * 32;
#pragma unroll
        for (int v = 0; v < 8; v++) {
          int kb = (v & 3) * 2 + hf * 8 + (v >> 2) * 16 + kk;
          ap[kh].u[v] = *(const unsigned*)&Ps[rowA * ST + kb];
        }
#pragma unroll
        for (int c = 0; c < 4; c++) {
          int nd = c * 16 + ln;
#pragma unroll
          for (int v = 0; v < 8; v++) {
            int kb = (v & 3) * 2 + hf * 8 + (v >> 2) * 16 + kk;
            bv[kh][c].u[v] = *(const unsigned*)&Vs[nd * ST + kb];
          }
        }
      }
#pragma unroll
      for (int kh = 0; kh < 2; kh++)
#pragma unroll
        for (int c = 0; c < 4; c++) o[c] = wmma_bf16(ap[kh], bv[kh][c], o[c]);
    }
  }

#pragma unroll
  for (int c = 0; c < 4; c++)
#pragma unroll
    for (int e = 0; e < 8; e++) {
      int m = wave * 16 + e + 8 * hf;
      int row = q0 + m;
      if (row < NT)
        outp[(size_t)(b * NT + row) * EDIM + h * DHEAD + c * 16 + ln] =
            f2bf(o[c][e] / l_run[e]);
    }
}

// ---------------------------------------------------------------------------
// LayerNorm over E=768, one row per 256-thread block; OutT = fp32 or bf16.
// ---------------------------------------------------------------------------
template <typename OutT>
__global__ __launch_bounds__(256) void ln_k(const float* __restrict__ X,
                                            const float* __restrict__ g,
                                            const float* __restrict__ bb,
                                            OutT* __restrict__ Y) {
  __shared__ float red[256];
  const int row = blockIdx.x, t = threadIdx.x;
  const float* xr = X + (size_t)row * EDIM;
  float v0 = xr[t], v1 = xr[t + 256], v2 = xr[t + 512];
  red[t] = v0 + v1 + v2;
  __syncthreads();
  for (int o = 128; o > 0; o >>= 1) {
    if (t < o) red[t] += red[t + o];
    __syncthreads();
  }
  float mean = red[0] * (1.f / EDIM);
  __syncthreads();
  float d0 = v0 - mean, d1 = v1 - mean, d2 = v2 - mean;
  red[t] = d0 * d0 + d1 * d1 + d2 * d2;
  __syncthreads();
  for (int o = 128; o > 0; o >>= 1) {
    if (t < o) red[t] += red[t + o];
    __syncthreads();
  }
  float rstd = rsqrtf(red[0] * (1.f / EDIM) + EPSLN);
  OutT* yr = Y + (size_t)row * EDIM;
  storeOut(&yr[t],       d0 * rstd * g[t]       + bb[t]);
  storeOut(&yr[t + 256], d1 * rstd * g[t + 256] + bb[t + 256]);
  storeOut(&yr[t + 512], d2 * rstd * g[t + 512] + bb[t + 512]);
}

// im2col: x[2,1,112,256,176] fp32 -> patches[2464,4096] bf16
__global__ void im2col_k(const float* __restrict__ x, bf16_t* __restrict__ p) {
  int i = blockIdx.x * 256 + threadIdx.x;
  if (i >= MROWS * KPATCH) return;
  int row = i >> 12, k = i & 4095;
  int b = row / NT, n = row % NT;
  int d = n / 176;
  int rr = n % 176;
  int hh = rr / 11, w = rr % 11;
  int pd = k >> 8, ph = (k >> 4) & 15, pw = k & 15;
  size_t xi = ((size_t)b * 112 + d * 16 + pd) * (256 * 176) +
              (size_t)(hh * 16 + ph) * 176 + (w * 16 + pw);
  p[i] = f2bf(x[xi]);
}

// fp32 [L][K][N] -> bf16 [L][N][K] (transpose + convert)
__global__ void cvtT_k(const float* __restrict__ W, bf16_t* __restrict__ Wt,
                       int K, int N, long total) {
  long i = (long)blockIdx.x * 256 + threadIdx.x;
  if (i >= total) return;
  long per = (long)K * N;
  int l = (int)(i / per);
  long r = i - (long)l * per;
  int n = (int)(r / K), k = (int)(r % K);
  Wt[i] = f2bf(W[(long)l * per + (long)k * N + n]);
}

// fp32 -> bf16, same layout
__global__ void cvt_k(const float* __restrict__ W, bf16_t* __restrict__ Wt,
                      long total) {
  long i = (long)blockIdx.x * 256 + threadIdx.x;
  if (i >= total) return;
  Wt[i] = f2bf(W[i]);
}

// tok += pos_emb broadcast over batch
__global__ void posadd_k(float* __restrict__ tok, const float* __restrict__ pos,
                         int total) {
  int i = blockIdx.x * 256 + threadIdx.x;
  if (i >= total) return;
  tok[i] += pos[i % (NT * EDIM)];
}

// ---------------------------------------------------------------------------
extern "C" void kernel_launch(void* const* d_in, const int* in_sizes, int n_in,
                              void* d_out, int out_size, void* d_ws,
                              size_t ws_size, hipStream_t stream) {
  (void)in_sizes; (void)n_in; (void)out_size; (void)ws_size;
  const float* x       = (const float*)d_in[0];
  const float* conv_w  = (const float*)d_in[1];
  const float* conv_b  = (const float*)d_in[2];
  const float* pos_emb = (const float*)d_in[3];
  const float* ln1_s   = (const float*)d_in[4];
  const float* ln1_b   = (const float*)d_in[5];
  const float* qkv_w   = (const float*)d_in[6];
  const float* qkv_b   = (const float*)d_in[7];
  const float* ao_w    = (const float*)d_in[8];
  const float* ao_b    = (const float*)d_in[9];
  const float* ln2_s   = (const float*)d_in[10];
  const float* ln2_b   = (const float*)d_in[11];
  const float* mlp_w1  = (const float*)d_in[12];
  const float* mlp_b1  = (const float*)d_in[13];
  const float* mlp_w2  = (const float*)d_in[14];
  const float* mlp_b2  = (const float*)d_in[15];
  const float* lnf_s   = (const float*)d_in[16];
  const float* lnf_b   = (const float*)d_in[17];
  float* out = (float*)d_out;

  char* wsb = (char*)d_ws;
  size_t off = 0;
  auto alloc = [&](size_t bytes) -> void* {
    void* p = (void*)(wsb + off);
    off += (bytes + 255) & ~(size_t)255;
    return p;
  };
  // bf16 weights (converted; [N][K] layout)
  bf16_t* convWb = (bf16_t*)alloc((size_t)EDIM * KPATCH * 2);          //  6.3MB
  bf16_t* qkvWT  = (bf16_t*)alloc((size_t)NLAYER * QKVD * EDIM * 2);   // 42.5MB
  bf16_t* aoWT   = (bf16_t*)alloc((size_t)NLAYER * EDIM * EDIM * 2);   // 14.2MB
  bf16_t* w1T    = (bf16_t*)alloc((size_t)NLAYER * MLPD * EDIM * 2);   // 56.6MB
  bf16_t* w2T    = (bf16_t*)alloc((size_t)NLAYER * EDIM * MLPD * 2);   // 56.6MB
  // activations
  bf16_t* patches = (bf16_t*)alloc((size_t)MROWS * KPATCH * 2);        // 20.2MB
  float*  tok     = (float*)alloc((size_t)MROWS * EDIM * 4);           //  7.6MB
  bf16_t* yb      = (bf16_t*)alloc((size_t)MROWS * EDIM * 2);          //  3.8MB
  bf16_t* qkv     = (bf16_t*)alloc((size_t)MROWS * QKVD * 2);          // 11.4MB
  bf16_t* attn_o  = (bf16_t*)alloc((size_t)MROWS * EDIM * 2);          //  3.8MB
  bf16_t* mlp_h   = (bf16_t*)alloc((size_t)MROWS * MLPD * 2);          // 15.1MB

  const int mtiles = (MROWS + 127) / 128;  // 20
  auto blocks = [](long total) { return (unsigned)((total + 255) / 256); };

  // weight conversion (conv_w is already [E][4096] == [N][K])
  cvt_k<<<blocks((long)EDIM * KPATCH), 256, 0, stream>>>(
      conv_w, convWb, (long)EDIM * KPATCH);
  cvtT_k<<<blocks((long)NLAYER * EDIM * QKVD), 256, 0, stream>>>(
      qkv_w, qkvWT, EDIM, QKVD, (long)NLAYER * EDIM * QKVD);
  cvtT_k<<<blocks((long)NLAYER * EDIM * EDIM), 256, 0, stream>>>(
      ao_w, aoWT, EDIM, EDIM, (long)NLAYER * EDIM * EDIM);
  cvtT_k<<<blocks((long)NLAYER * EDIM * MLPD), 256, 0, stream>>>(
      mlp_w1, w1T, EDIM, MLPD, (long)NLAYER * EDIM * MLPD);
  cvtT_k<<<blocks((long)NLAYER * MLPD * EDIM), 256, 0, stream>>>(
      mlp_w2, w2T, MLPD, EDIM, (long)NLAYER * MLPD * EDIM);

  im2col_k<<<blocks((long)MROWS * KPATCH), 256, 0, stream>>>(x, patches);

  // patch embedding: tok = patches @ conv_w^T + conv_b
  gemm_k<0, float><<<dim3(EDIM / 128, mtiles), 256, 0, stream>>>(
      patches, convWb, conv_b, nullptr, tok, MROWS, EDIM, KPATCH);
  posadd_k<<<blocks((long)MROWS * EDIM), 256, 0, stream>>>(
      tok, pos_emb, MROWS * EDIM);

  for (int i = 0; i < NLAYER; i++) {
    ln_k<bf16_t><<<MROWS, 256, 0, stream>>>(tok, ln1_s + (size_t)i * EDIM,
                                            ln1_b + (size_t)i * EDIM, yb);
    gemm_k<0, bf16_t><<<dim3(QKVD / 128, mtiles), 256, 0, stream>>>(
        yb, qkvWT + (size_t)i * QKVD * EDIM, qkv_b + (size_t)i * QKVD, nullptr,
        qkv, MROWS, QKVD, EDIM);
    attn_k<<<dim3((NT + 63) / 64, BATCH * HEADS), 128, 0, stream>>>(qkv,
                                                                    attn_o);
    gemm_k<2, float><<<dim3(EDIM / 128, mtiles), 256, 0, stream>>>(
        attn_o, aoWT + (size_t)i * EDIM * EDIM, ao_b + (size_t)i * EDIM, tok,
        tok, MROWS, EDIM, EDIM);
    ln_k<bf16_t><<<MROWS, 256, 0, stream>>>(tok, ln2_s + (size_t)i * EDIM,
                                            ln2_b + (size_t)i * EDIM, yb);
    gemm_k<1, bf16_t><<<dim3(MLPD / 128, mtiles), 256, 0, stream>>>(
        yb, w1T + (size_t)i * MLPD * EDIM, mlp_b1 + (size_t)i * MLPD, nullptr,
        mlp_h, MROWS, MLPD, EDIM);
    gemm_k<2, float><<<dim3(EDIM / 128, mtiles), 256, 0, stream>>>(
        mlp_h, w2T + (size_t)i * EDIM * MLPD, mlp_b2 + (size_t)i * EDIM, tok,
        tok, MROWS, EDIM, MLPD);
  }

  ln_k<float><<<MROWS, 256, 0, stream>>>(tok, lnf_s, lnf_b, out);
}
